// CausalSelfAttention_9826885173455
// MI455X (gfx1250) — compile-verified, matrix operand order checked by reading, not run on hardware
//
#include <hip/hip_runtime.h>
#include <hip/hip_bf16.h>
#include <stdint.h>

typedef __attribute__((ext_vector_type(8)))  __bf16 v8bf;
typedef __attribute__((ext_vector_type(16))) __bf16 v16bf;
typedef __attribute__((ext_vector_type(8)))  float  v8f;

#define DEVINL __device__ __forceinline__

static constexpr int B_ = 2, T_ = 2048, C_ = 1024, H_ = 16, D_ = 64;
static constexpr int M_ = B_ * T_;      // 4096 rows of x / attn
static constexpr int NQKV_ = 3 * C_;    // 3072

// ---------- helpers ----------

DEVINL unsigned short f32_to_bf16(float f) {
  unsigned u = __float_as_uint(f);
  unsigned r = u + 0x7FFFu + ((u >> 16) & 1u);  // RNE
  return (unsigned short)(r >> 16);
}

// Load a 16x32 bf16 fragment (A layout; identical layout serves B when the
// source matrix is [N,K] row-major). ISA 7.12.2 "16-bit A-Matrix 16x32":
// lanes 0-15: row=row0+lane, K blocks {0..7, 16..23}; lanes 16-31: same rows,
// K blocks {8..15, 24..31}. Two 16-byte loads per lane.
DEVINL v16bf load_frag(const unsigned short* base, int row0, int k0, int ld) {
  int lane = threadIdx.x & 31;
  int half = lane >> 4;
  const unsigned short* p =
      base + (size_t)(row0 + (lane & 15)) * ld + k0 + half * 8;
  v8bf lo = *reinterpret_cast<const v8bf*>(p);
  v8bf hi = *reinterpret_cast<const v8bf*>(p + 16);
  return __builtin_shufflevector(lo, hi,
      0, 1, 2, 3, 4, 5, 6, 7, 8, 9, 10, 11, 12, 13, 14, 15);
}

DEVINL v8f wmma_bf16(v16bf a, v16bf b, v8f c) {
  return __builtin_amdgcn_wmma_f32_16x16x32_bf16(
      false, a, false, b, (short)0, c, false, false);
}

// Reductions across lanes 0..15 / 16..31 (xor masks < 16 keep halves apart).
// ds_swizzle group-of-32 imm: {xor[14:10], or[9:5]=0, and[4:0]=0x1F}.
DEVINL float lane16_max(float x) {
  x = fmaxf(x, __uint_as_float(__builtin_amdgcn_ds_swizzle(__float_as_uint(x), 0x041F)));
  x = fmaxf(x, __uint_as_float(__builtin_amdgcn_ds_swizzle(__float_as_uint(x), 0x081F)));
  x = fmaxf(x, __uint_as_float(__builtin_amdgcn_ds_swizzle(__float_as_uint(x), 0x101F)));
  x = fmaxf(x, __uint_as_float(__builtin_amdgcn_ds_swizzle(__float_as_uint(x), 0x201F)));
  return x;
}
DEVINL float lane16_sum(float x) {
  x += __uint_as_float(__builtin_amdgcn_ds_swizzle(__float_as_uint(x), 0x041F));
  x += __uint_as_float(__builtin_amdgcn_ds_swizzle(__float_as_uint(x), 0x081F));
  x += __uint_as_float(__builtin_amdgcn_ds_swizzle(__float_as_uint(x), 0x101F));
  x += __uint_as_float(__builtin_amdgcn_ds_swizzle(__float_as_uint(x), 0x201F));
  return x;
}

// ---------- kernels ----------

__global__ void csa_cast_f32_bf16(const float* __restrict__ in,
                                  unsigned short* __restrict__ out, int n) {
  int i = blockIdx.x * blockDim.x + threadIdx.x;
  int stride = gridDim.x * blockDim.x;
  for (; i < n; i += stride) out[i] = f32_to_bf16(in[i]);
}

// qkv = x @ w_qkv^T ; split + relayout epilogue.
// Block: 256 threads = 8 waves in 2x4; wave tile 32x64 (2 A-frags x 4 B-frags
// -> 8 WMMA per 6 fragment loads); block tile 64x256.
__global__ void __launch_bounds__(256) csa_qkv_gemm(
    const unsigned short* __restrict__ X,   // [M_, C_] bf16
    const unsigned short* __restrict__ W,   // [3C_, C_] bf16 (N-major)
    unsigned short* __restrict__ Q,         // [B,H,T,D] pre-scaled
    unsigned short* __restrict__ K,         // [B,H,T,D]
    unsigned short* __restrict__ VT)        // [B,H,D,T]
{
  const int tid = threadIdx.x;
  const int wave = tid >> 5;
  const int lane = tid & 31;
  const int half = lane >> 4;
  const int ln = lane & 15;
  const int m0 = blockIdx.y * 64 + (wave >> 2) * 32;
  const int n0 = blockIdx.x * 256 + (wave & 3) * 64;

  v8f acc[2][4];
#pragma unroll
  for (int i = 0; i < 2; ++i)
#pragma unroll
    for (int j = 0; j < 4; ++j) acc[i][j] = (v8f){};

  for (int k0 = 0; k0 < C_; k0 += 32) {
    if (k0 + 32 < C_) {
      __builtin_prefetch(X + (size_t)m0 * C_ + k0 + 32, 0, 1);
      __builtin_prefetch(W + (size_t)n0 * C_ + k0 + 32, 0, 1);
    }
    v16bf a0 = load_frag(X, m0,      k0, C_);
    v16bf a1 = load_frag(X, m0 + 16, k0, C_);
    v16bf b0 = load_frag(W, n0,      k0, C_);
    v16bf b1 = load_frag(W, n0 + 16, k0, C_);
    v16bf b2 = load_frag(W, n0 + 32, k0, C_);
    v16bf b3 = load_frag(W, n0 + 48, k0, C_);
    acc[0][0] = wmma_bf16(a0, b0, acc[0][0]);
    acc[0][1] = wmma_bf16(a0, b1, acc[0][1]);
    acc[0][2] = wmma_bf16(a0, b2, acc[0][2]);
    acc[0][3] = wmma_bf16(a0, b3, acc[0][3]);
    acc[1][0] = wmma_bf16(a1, b0, acc[1][0]);
    acc[1][1] = wmma_bf16(a1, b1, acc[1][1]);
    acc[1][2] = wmma_bf16(a1, b2, acc[1][2]);
    acc[1][3] = wmma_bf16(a1, b3, acc[1][3]);
  }

#pragma unroll
  for (int mi = 0; mi < 2; ++mi)
#pragma unroll
    for (int ni = 0; ni < 4; ++ni)
#pragma unroll
      for (int r = 0; r < 8; ++r) {
        int t = m0 + mi * 16 + r + half * 8;   // global row (b*T + t)
        int colg = n0 + ni * 16 + ln;          // global column in [0,3072)
        float val = acc[mi][ni][r];
        int b = t >> 11, tt = t & (T_ - 1);
        int sel = colg >> 10, c = colg & (C_ - 1);
        int h = c >> 6, d = c & 63;
        size_t bh = (size_t)(b * H_ + h);
        if (sel == 0)
          Q[(bh * T_ + tt) * D_ + d] = f32_to_bf16(val * 0.125f); // 1/sqrt(64)
        else if (sel == 1)
          K[(bh * T_ + tt) * D_ + d] = f32_to_bf16(val);
        else
          VT[(bh * D_ + d) * T_ + tt] = f32_to_bf16(val);
      }
}

// Flash attention: one block = one (b,h) x 64-query tile; 4 waves,
// each wave owns a 16-row Q strip and its full softmax state.
__global__ void __launch_bounds__(128) csa_flash_attn(
    const unsigned short* __restrict__ Q,    // [B,H,T,D] pre-scaled bf16
    const unsigned short* __restrict__ Kb,   // [B,H,T,D]
    const unsigned short* __restrict__ VT,   // [B,H,D,T]
    unsigned short* __restrict__ A)          // [B,T,C] bf16
{
  __shared__ unsigned short lds_p[4 * 16 * 64];  // per-wave P strip, 8 KB

  const int tid = threadIdx.x;
  const int wave = tid >> 5;
  const int lane = tid & 31;
  const int half = lane >> 4;
  const int ln = lane & 15;
  const int bh = blockIdx.y;
  const int b = bh >> 4, h = bh & 15;
  const int q0 = blockIdx.x * 64;
  const int qr = q0 + wave * 16;  // first q row of this wave's strip

  const unsigned short* qh = Q  + (size_t)bh * T_ * D_;
  const unsigned short* kh = Kb + (size_t)bh * T_ * D_;
  const unsigned short* vh = VT + (size_t)bh * D_ * T_;

  v16bf aq0 = load_frag(qh, qr, 0, D_);
  v16bf aq1 = load_frag(qh, qr, 32, D_);

  v8f o[4];
#pragma unroll
  for (int i = 0; i < 4; ++i) o[i] = (v8f){};
  float mrun[8], lrun[8];
#pragma unroll
  for (int r = 0; r < 8; ++r) { mrun[r] = -1.0e30f; lrun[r] = 0.0f; }

  unsigned short* lp = lds_p + wave * 16 * 64;

  for (int kt = 0; kt <= q0; kt += 64) {
    // S strip (16x64) = Q_strip @ K_tile^T
    v8f s[4];
#pragma unroll
    for (int tn = 0; tn < 4; ++tn) s[tn] = (v8f){};
#pragma unroll
    for (int tn = 0; tn < 4; ++tn) {
      v16bf bk0 = load_frag(kh, kt + tn * 16, 0, D_);
      v16bf bk1 = load_frag(kh, kt + tn * 16, 32, D_);
      s[tn] = wmma_bf16(aq0, bk0, s[tn]);
      s[tn] = wmma_bf16(aq1, bk1, s[tn]);
    }
    // causal mask + online softmax per row
#pragma unroll
    for (int r = 0; r < 8; ++r) {
      int qi = qr + r + half * 8;
#pragma unroll
      for (int tn = 0; tn < 4; ++tn) {
        int kj = kt + tn * 16 + ln;
        if (kj > qi) s[tn][r] = -1.0e30f;
      }
      float rm = fmaxf(fmaxf(s[0][r], s[1][r]), fmaxf(s[2][r], s[3][r]));
      rm = lane16_max(rm);
      float mn = fmaxf(mrun[r], rm);
      float corr = __expf(mrun[r] - mn);
      float rs = 0.0f;
#pragma unroll
      for (int tn = 0; tn < 4; ++tn) {
        float p = __expf(s[tn][r] - mn);
        s[tn][r] = p;
        rs += p;
      }
      rs = lane16_sum(rs);
      lrun[r] = lrun[r] * corr + rs;
      mrun[r] = mn;
#pragma unroll
      for (int dn = 0; dn < 4; ++dn) o[dn][r] = o[dn][r] * corr;
    }
    // P (C layout) -> per-wave LDS strip as bf16
#pragma unroll
    for (int tn = 0; tn < 4; ++tn)
#pragma unroll
      for (int r = 0; r < 8; ++r)
        lp[(r + half * 8) * 64 + tn * 16 + ln] = f32_to_bf16(s[tn][r]);
    asm volatile("s_wait_dscnt 0" ::: "memory");  // wave-local LDS RAW fence
    // Re-read P as A fragments; O += P @ V (V^T gives contiguous B frags)
    v16bf p0 = load_frag(lp, 0, 0, 64);
    v16bf p1 = load_frag(lp, 0, 32, 64);
#pragma unroll
    for (int dn = 0; dn < 4; ++dn) {
      v16bf bv0 = load_frag(vh, dn * 16, kt, T_);
      v16bf bv1 = load_frag(vh, dn * 16, kt + 32, T_);
      o[dn] = wmma_bf16(p0, bv0, o[dn]);
      o[dn] = wmma_bf16(p1, bv1, o[dn]);
    }
  }

  // normalize and store attn [B,T,C]
#pragma unroll
  for (int r = 0; r < 8; ++r) {
    float inv = 1.0f / lrun[r];
    int t = qr + r + half * 8;
#pragma unroll
    for (int dn = 0; dn < 4; ++dn) {
      int d = dn * 16 + ln;
      A[((size_t)b * T_ + t) * C_ + h * D_ + d] = f32_to_bf16(o[dn][r] * inv);
    }
  }
}

// out = attn @ w_out^T  -> fp32. Same 32x64 wave tile / 64x256 block tile.
__global__ void __launch_bounds__(256) csa_out_gemm(
    const unsigned short* __restrict__ Ain,  // [M_, C_] bf16
    const unsigned short* __restrict__ W,    // [C_, C_] bf16
    float* __restrict__ Out)                 // [M_, C_] f32
{
  const int tid = threadIdx.x;
  const int wave = tid >> 5;
  const int lane = tid & 31;
  const int half = lane >> 4;
  const int ln = lane & 15;
  const int m0 = blockIdx.y * 64 + (wave >> 2) * 32;
  const int n0 = blockIdx.x * 256 + (wave & 3) * 64;

  v8f acc[2][4];
#pragma unroll
  for (int i = 0; i < 2; ++i)
#pragma unroll
    for (int j = 0; j < 4; ++j) acc[i][j] = (v8f){};

  for (int k0 = 0; k0 < C_; k0 += 32) {
    if (k0 + 32 < C_) {
      __builtin_prefetch(Ain + (size_t)m0 * C_ + k0 + 32, 0, 1);
      __builtin_prefetch(W + (size_t)n0 * C_ + k0 + 32, 0, 1);
    }
    v16bf a0 = load_frag(Ain, m0,      k0, C_);
    v16bf a1 = load_frag(Ain, m0 + 16, k0, C_);
    v16bf b0 = load_frag(W, n0,      k0, C_);
    v16bf b1 = load_frag(W, n0 + 16, k0, C_);
    v16bf b2 = load_frag(W, n0 + 32, k0, C_);
    v16bf b3 = load_frag(W, n0 + 48, k0, C_);
    acc[0][0] = wmma_bf16(a0, b0, acc[0][0]);
    acc[0][1] = wmma_bf16(a0, b1, acc[0][1]);
    acc[0][2] = wmma_bf16(a0, b2, acc[0][2]);
    acc[0][3] = wmma_bf16(a0, b3, acc[0][3]);
    acc[1][0] = wmma_bf16(a1, b0, acc[1][0]);
    acc[1][1] = wmma_bf16(a1, b1, acc[1][1]);
    acc[1][2] = wmma_bf16(a1, b2, acc[1][2]);
    acc[1][3] = wmma_bf16(a1, b3, acc[1][3]);
  }

#pragma unroll
  for (int mi = 0; mi < 2; ++mi)
#pragma unroll
    for (int ni = 0; ni < 4; ++ni)
#pragma unroll
      for (int r = 0; r < 8; ++r) {
        int t = m0 + mi * 16 + r + half * 8;
        int colg = n0 + ni * 16 + ln;
        Out[(size_t)t * C_ + colg] = acc[mi][ni][r];
      }
}

// ---------- launch ----------

extern "C" void kernel_launch(void* const* d_in, const int* in_sizes, int n_in,
                              void* d_out, int out_size, void* d_ws, size_t ws_size,
                              hipStream_t stream) {
  const float* x     = (const float*)d_in[0];
  const float* w_qkv = (const float*)d_in[1];
  const float* w_out = (const float*)d_in[2];
  float* out = (float*)d_out;

  char* ws = (char*)d_ws;
  const size_t MiB = 1ull << 20;
  unsigned short* xb    = (unsigned short*)(ws + 0 * MiB);   // 8 MiB
  unsigned short* wqkvb = (unsigned short*)(ws + 8 * MiB);   // 6 MiB
  unsigned short* woutb = (unsigned short*)(ws + 14 * MiB);  // 2 MiB
  unsigned short* qb    = (unsigned short*)(ws + 16 * MiB);  // 8 MiB
  unsigned short* kb    = (unsigned short*)(ws + 24 * MiB);  // 8 MiB
  unsigned short* vtb   = (unsigned short*)(ws + 32 * MiB);  // 8 MiB
  unsigned short* ab    = (unsigned short*)(ws + 40 * MiB);  // 8 MiB

  csa_cast_f32_bf16<<<1024, 256, 0, stream>>>(x, xb, M_ * C_);
  csa_cast_f32_bf16<<<1024, 256, 0, stream>>>(w_qkv, wqkvb, NQKV_ * C_);
  csa_cast_f32_bf16<<<1024, 256, 0, stream>>>(w_out, woutb, C_ * C_);

  csa_qkv_gemm<<<dim3(NQKV_ / 256, M_ / 64), 256, 0, stream>>>(
      xb, wqkvb, qb, kb, vtb);

  csa_flash_attn<<<dim3(T_ / 64, B_ * H_), 128, 0, stream>>>(
      qb, kb, vtb, ab);

  csa_out_gemm<<<dim3(C_ / 256, M_ / 64), 256, 0, stream>>>(
      ab, woutb, out);
}
